// GAT3View_MoE_71365176590650
// MI455X (gfx1250) — matrix operand order repaired and checked by reading, not compiled, not run on hardware
//
#include <hip/hip_runtime.h>
#include <cstddef>
#include <cstdint>

// ---------------------------------------------------------------------------
// Model constants (from the reference)
// ---------------------------------------------------------------------------
#define IN_DIM  768
#define META    6
#define XDIM    (IN_DIM + META)   // 774
#define XPAD    776               // padded x row stride (halves, 16B aligned)
#define HID     256
#define OUTD    128
#define NEXP    4
#define NHEAD   4
#define NLAYER  4
#define HF      (4 * OUTD)        // 512

#define F_RELU  1
#define F_ACCUM 2

typedef __attribute__((ext_vector_type(16))) _Float16 v16h;
typedef __attribute__((ext_vector_type(8)))  _Float16 v8h;
typedef __attribute__((ext_vector_type(8)))  float    v8f;

typedef __attribute__((address_space(3))) _Float16 lds_h_t;

// ---------------------------------------------------------------------------
// WMMA GEMM on f16 operands:
//   C[m,n] (op)= act( A[m,:K] dot W[n,:K] + bias[n] ) * rs[m]
//   A: [M,K] f16, row stride lda (halves, multiple of 8)
//   W: [N,K] f16, row stride ldw (halves, multiple of 8)
//   C   (nullable): f32 out, stride ldc;  C16 (nullable): f16 out, stride ldc
//   flags: F_RELU ; F_ACCUM (f32 C += v)
//
// One wave computes a 16x32 output slab (two WMMA tiles sharing the A frag).
// Interior tiles stream K in steps of 32 via GLOBAL_LOAD_ASYNC_TO_LDS_B128
// into DOUBLE-BUFFERED LDS stages: stage i+1's 6 async chunks are issued
// before stage i is consumed, and s_wait_asynccnt 6 overlaps the DMA with the
// WMMA work (async loads complete in order). Edge tiles / K tail use a
// predicated scalar path.
// ---------------------------------------------------------------------------
__global__ __launch_bounds__(256)
void k_gemm(const _Float16* __restrict__ A, int lda,
            const _Float16* __restrict__ W, int ldw,
            const float* __restrict__ bias,
            const float* __restrict__ rowscale, int rs_stride,
            float* __restrict__ C, _Float16* __restrict__ C16, int ldc,
            int M, int N, int K, int flags)
{
    // per-wave double-buffered stages: A = 2 x 16x32, B = 2 x 32x32 (halves)
    __shared__ __attribute__((aligned(32))) _Float16 sA[8][2 * 16 * 32];
    __shared__ __attribute__((aligned(32))) _Float16 sB[8][2 * 32 * 32];

    const int wave = threadIdx.x >> 5;
    const int lane = threadIdx.x & 31;
    const int tiles_m  = (M + 15) >> 4;
    const int tiles_n2 = (N + 31) >> 5;
    const int tile = blockIdx.x * 8 + wave;
    if (tile >= tiles_m * tiles_n2) return;          // wave-uniform branch
    const int m0 = (tile / tiles_n2) << 4;
    const int n0 = (tile % tiles_n2) << 5;

    _Float16* a_sh = sA[wave];
    _Float16* b_sh = sB[wave];

    v8f acc0 = {0.f, 0.f, 0.f, 0.f, 0.f, 0.f, 0.f, 0.f};
    v8f acc1 = {0.f, 0.f, 0.f, 0.f, 0.f, 0.f, 0.f, 0.f};

    const int mh = lane >> 4;          // half-wave select
    const int ml = lane & 15;          // row (A) / col (B) within tile

    const bool interior = (m0 + 16 <= M) && (n0 + 32 <= N);

    int k0 = 0;
    const int nk = K >> 5;             // full K-steps
    if (interior && nk > 0) {
        // LDS byte offsets of this wave's staging buffers (addrspace(3) view)
        const unsigned aoff = (unsigned)(size_t)(lds_h_t*)a_sh;
        const unsigned boff = (unsigned)(size_t)(lds_h_t*)b_sh;
        // chunk layout: 16B chunks, row = chunk>>2, blk = chunk&3
        const unsigned crow = (unsigned)(lane >> 2);
        const unsigned cblk = (unsigned)(lane & 3);
        const unsigned ldsA0 = aoff + (unsigned)lane * 16u;
        const unsigned ldsB0 = boff + (unsigned)lane * 16u;
        unsigned voffA0 = crow * (unsigned)lda * 2u + cblk * 16u;
        unsigned voffA1 = (crow + 8u) * (unsigned)lda * 2u + cblk * 16u;
        unsigned voffB0 = crow * (unsigned)ldw * 2u + cblk * 16u;
        unsigned voffB1 = (crow + 8u)  * (unsigned)ldw * 2u + cblk * 16u;
        unsigned voffB2 = (crow + 16u) * (unsigned)ldw * 2u + cblk * 16u;
        unsigned voffB3 = (crow + 24u) * (unsigned)ldw * 2u + cblk * 16u;
        const unsigned long long baseA = (unsigned long long)(A + (size_t)m0 * lda);
        const unsigned long long baseB = (unsigned long long)(W + (size_t)n0 * ldw);

        // issue one 6-chunk async group into stage st (0/1), advance voffsets
        auto issue = [&](unsigned st) {
            const unsigned sa = ldsA0 + st * 1024u;   // A stage = 1KB
            const unsigned sb = ldsB0 + st * 2048u;   // B stage = 2KB
            asm volatile("global_load_async_to_lds_b128 %0, %1, %2"
                         :: "v"(sa),          "v"(voffA0), "s"(baseA) : "memory");
            asm volatile("global_load_async_to_lds_b128 %0, %1, %2"
                         :: "v"(sa + 512u),   "v"(voffA1), "s"(baseA) : "memory");
            asm volatile("global_load_async_to_lds_b128 %0, %1, %2"
                         :: "v"(sb),          "v"(voffB0), "s"(baseB) : "memory");
            asm volatile("global_load_async_to_lds_b128 %0, %1, %2"
                         :: "v"(sb + 512u),   "v"(voffB1), "s"(baseB) : "memory");
            asm volatile("global_load_async_to_lds_b128 %0, %1, %2"
                         :: "v"(sb + 1024u),  "v"(voffB2), "s"(baseB) : "memory");
            asm volatile("global_load_async_to_lds_b128 %0, %1, %2"
                         :: "v"(sb + 1536u),  "v"(voffB3), "s"(baseB) : "memory");
            voffA0 += 64u; voffA1 += 64u;
            voffB0 += 64u; voffB1 += 64u; voffB2 += 64u; voffB3 += 64u;
        };

        issue(0u);                                    // prologue
        for (int i = 0; i < nk; ++i) {
            const bool more = (i + 1) < nk;
            if (more) {
                issue((unsigned)((i + 1) & 1));       // prefetch next stage
                asm volatile("s_wait_asynccnt 6" ::: "memory");  // prev group done
            } else {
                asm volatile("s_wait_asynccnt 0" ::: "memory");
            }
            const int st = i & 1;
            const _Float16* ast = a_sh + st * 512;
            const _Float16* bst = b_sh + st * 1024;

            // A fragment (16-bit A 16x32 layout)
            v8h a_lo = *(const v8h*)(ast + ml * 32 + mh * 16);
            v8h a_hi = *(const v8h*)(ast + ml * 32 + mh * 16 + 8);
            v16h af = __builtin_shufflevector(a_lo, a_hi,
                0, 1, 2, 3, 4, 5, 6, 7, 8, 9, 10, 11, 12, 13, 14, 15);
            // B fragments (16-bit B 32x16): contiguous 16-half block per lane
            v16h b0 = *(const v16h*)(bst + ml * 32 + mh * 16);
            v16h b1 = *(const v16h*)(bst + (ml + 16) * 32 + mh * 16);

            acc0 = __builtin_amdgcn_wmma_f32_16x16x32_f16(
                false, af, false, b0, (short)0, acc0, false, false);
            acc1 = __builtin_amdgcn_wmma_f32_16x16x32_f16(
                false, af, false, b1, (short)0, acc1, false, false);
        }
        k0 = nk << 5;
    }
    // ---- predicated path: edge tiles and K tail ----
    for (; k0 < K; k0 += 32) {
        const int kk = k0 + lane;
        const bool kok = kk < K;
        for (int i = 0; i < 16; ++i) {
            _Float16 av = ((m0 + i) < M && kok) ? A[(size_t)(m0 + i) * lda + kk]
                                                : (_Float16)0.f;
            a_sh[i * 32 + lane] = av;
        }
        for (int i = 0; i < 32; ++i) {
            _Float16 bv = ((n0 + i) < N && kok) ? W[(size_t)(n0 + i) * ldw + kk]
                                                : (_Float16)0.f;
            b_sh[i * 32 + lane] = bv;
        }
        asm volatile("s_wait_dscnt 0" ::: "memory");

        v8h a_lo = *(const v8h*)(a_sh + ml * 32 + mh * 16);
        v8h a_hi = *(const v8h*)(a_sh + ml * 32 + mh * 16 + 8);
        v16h af = __builtin_shufflevector(a_lo, a_hi,
            0, 1, 2, 3, 4, 5, 6, 7, 8, 9, 10, 11, 12, 13, 14, 15);
        v16h b0 = *(const v16h*)(b_sh + ml * 32 + mh * 16);
        v16h b1 = *(const v16h*)(b_sh + (ml + 16) * 32 + mh * 16);

        acc0 = __builtin_amdgcn_wmma_f32_16x16x32_f16(
            false, af, false, b0, (short)0, acc0, false, false);
        acc1 = __builtin_amdgcn_wmma_f32_16x16x32_f16(
            false, af, false, b1, (short)0, acc1, false, false);
    }

    // ---- epilogue: elem r -> row m0 + r + 8*mh ; cols n0+ml and n0+16+ml ----
    const int na = n0 + ml;
    const int nb = n0 + 16 + ml;
    const float bva = (bias && na < N) ? bias[na] : 0.0f;
    const float bvb = (bias && nb < N) ? bias[nb] : 0.0f;
    for (int r = 0; r < 8; ++r) {
        const int m = m0 + r + mh * 8;
        if (m >= M) continue;
        const float rs = rowscale ? rowscale[(size_t)m * rs_stride] : 1.0f;
        if (na < N) {
            float v = acc0[r] + bva;
            if (flags & F_RELU) v = fmaxf(v, 0.0f);
            v *= rs;
            const size_t idx = (size_t)m * ldc + na;
            if (C)  { if (flags & F_ACCUM) C[idx] += v; else C[idx] = v; }
            if (C16) C16[idx] = (_Float16)v;
        }
        if (nb < N) {
            float v = acc1[r] + bvb;
            if (flags & F_RELU) v = fmaxf(v, 0.0f);
            v *= rs;
            const size_t idx = (size_t)m * ldc + nb;
            if (C)  { if (flags & F_ACCUM) C[idx] += v; else C[idx] = v; }
            if (C16) C16[idx] = (_Float16)v;
        }
    }
}

// ---------------------------------------------------------------------------
// Conversion / utility kernels
// ---------------------------------------------------------------------------
__global__ void k_cvt(const float* __restrict__ s, _Float16* __restrict__ d, int n) {
    int i = blockIdx.x * blockDim.x + threadIdx.x;
    if (i < n) d[i] = (_Float16)s[i];
}
// [Nrows,K] f32 -> [Nrows,Kp] f16 with zero-padded columns
__global__ void k_cvt_pad(const float* __restrict__ s, _Float16* __restrict__ d,
                          int Nrows, int K, int Kp) {
    int t = blockIdx.x * blockDim.x + threadIdx.x;
    if (t >= Nrows * Kp) return;
    int r = t / Kp, c = t % Kp;
    d[t] = (c < K) ? (_Float16)s[(size_t)r * K + c] : (_Float16)0.f;
}
__global__ void k_zero(float* p, int n) {
    int i = blockIdx.x * blockDim.x + threadIdx.x;
    if (i < n) p[i] = 0.0f;
}
__global__ void k_fillu(unsigned* p, unsigned v, int n) {
    int i = blockIdx.x * blockDim.x + threadIdx.x;
    if (i < n) p[i] = v;
}

__device__ __forceinline__ unsigned ford(float f) {
    unsigned u = __float_as_uint(f);
    return (u & 0x80000000u) ? ~u : (u | 0x80000000u);
}
__device__ __forceinline__ float funord(unsigned u) {
    return (u & 0x80000000u) ? __uint_as_float(u & 0x7fffffffu)
                             : __uint_as_float(~u);
}

// per-node attention coefficients: asrc[i] = <h[i,:], a_src>, adst likewise
__global__ void k_alpha(const float* __restrict__ h,
                        const float* __restrict__ aw_s,
                        const float* __restrict__ aw_d,
                        float* asrc, float* adst, int Nn)
{
    int i = blockIdx.x * blockDim.x + threadIdx.x;
    if (i >= Nn) return;
    const float* row = h + (size_t)i * HID;
    float s = 0.f, d = 0.f;
    for (int j = 0; j < HID; ++j) { float v = row[j]; s += v * aw_s[j]; d += v * aw_d[j]; }
    asrc[i] = s; adst[i] = d;
}

__global__ void k_edge_max(const int* __restrict__ src, const int* __restrict__ dst,
                           int E, int E2,
                           const float* __restrict__ asrc, const float* __restrict__ adst,
                           float* __restrict__ ebuf, unsigned* __restrict__ mmax)
{
    int e = blockIdx.x * blockDim.x + threadIdx.x;
    if (e >= E2) return;
    int s = (e < E) ? src[e] : (e - E);
    int d = (e < E) ? dst[e] : (e - E);
    float x = asrc[s] + adst[d];
    x = (x > 0.f) ? x : 0.2f * x;              // leaky_relu(0.2)
    ebuf[e] = x;
    atomicMax(&mmax[d], ford(x));
}

__global__ void k_edge_exp(const int* __restrict__ src, const int* __restrict__ dst,
                           int E, int E2,
                           float* __restrict__ ebuf,
                           const unsigned* __restrict__ mmax,
                           float* __restrict__ zsum)
{
    int e = blockIdx.x * blockDim.x + threadIdx.x;
    if (e >= E2) return;
    int d = (e < E) ? dst[e] : (e - E);
    float p = expf(ebuf[e] - funord(mmax[d]));
    ebuf[e] = p;
    atomicAdd(&zsum[d], p);
}

// out[dst,:] += (p/z[dst]) * hW[src,:]   (4 features per thread)
__global__ void k_edge_scatter(const int* __restrict__ src, const int* __restrict__ dst,
                               int E, int E2,
                               const float* __restrict__ ebuf,
                               const float* __restrict__ zsum,
                               const float* __restrict__ hW,
                               float* __restrict__ out)
{
    int t = blockIdx.x * blockDim.x + threadIdx.x;
    int e = t >> 6;
    if (e >= E2) return;
    int f4 = (t & 63) << 2;
    int s = (e < E) ? src[e] : (e - E);
    int d = (e < E) ? dst[e] : (e - E);
    float coef = ebuf[e] / zsum[d];
    const float* hs = hW + (size_t)s * HID + f4;
    float* od = out + (size_t)d * HID + f4;
    atomicAdd(&od[0], coef * hs[0]);
    atomicAdd(&od[1], coef * hs[1]);
    atomicAdd(&od[2], coef * hs[2]);
    atomicAdd(&od[3], coef * hs[3]);
}

__global__ void k_bias_act(float* __restrict__ h, const float* __restrict__ bias,
                           int total, int nf, int relu)
{
    int t = blockIdx.x * blockDim.x + threadIdx.x;
    if (t >= total) return;
    float v = h[t] + bias[t % nf];
    if (relu) v = fmaxf(v, 0.0f);
    h[t] = v;
}

// ---------------------------------------------------------------------------
// MoE noisy gating (NEXP==4, TOPK==2) on f16 inputs
// ---------------------------------------------------------------------------
__global__ void k_moe_gate(const _Float16* __restrict__ xx,  // [B,128] f16
                           const float* __restrict__ eps,    // [B,4]
                           const float* __restrict__ gate_w, // [4,128]
                           const float* __restrict__ noise_w,// [1,128]
                           float* __restrict__ wout,         // [B,4]
                           float* __restrict__ colsum,       // [4]
                           int B)
{
    int b = blockIdx.x * blockDim.x + threadIdx.x;
    if (b >= B) return;
    const _Float16* xr = xx + (size_t)b * OUTD;
    float g0 = 0.f, g1 = 0.f, g2 = 0.f, g3 = 0.f, nn = 0.f;
    for (int j = 0; j < OUTD; ++j) {
        float v = (float)xr[j];
        g0 += v * gate_w[j];
        g1 += v * gate_w[OUTD + j];
        g2 += v * gate_w[2 * OUTD + j];
        g3 += v * gate_w[3 * OUTD + j];
        nn += v * noise_w[j];
    }
    float sp = (nn > 20.f) ? nn : log1pf(expf(nn));
    float h[4];
    h[0] = g0 + eps[b * 4 + 0] * sp;
    h[1] = g1 + eps[b * 4 + 1] * sp;
    h[2] = g2 + eps[b * 4 + 2] * sp;
    h[3] = g3 + eps[b * 4 + 3] * sp;

    int i1 = 0;
    for (int e = 1; e < 4; ++e) if (h[e] > h[i1]) i1 = e;
    int i2 = -1;
    for (int e = 0; e < 4; ++e) {
        if (e == i1) continue;
        if (i2 < 0 || h[e] > h[i2]) i2 = e;
    }
    float m  = fmaxf(h[i1], h[i2]);
    float p1 = expf(h[i1] - m), p2 = expf(h[i2] - m);
    float z  = p1 + p2;
    float L1 = p1 / z, L2 = p2 / z;

    float w[4] = {0.f, 0.f, 0.f, 0.f};
    w[i1] = L1; w[i2] = L2;
    wout[b * 4 + 0] = w[0]; wout[b * 4 + 1] = w[1];
    wout[b * 4 + 2] = w[2]; wout[b * 4 + 3] = w[3];
    atomicAdd(&colsum[i1], L1);
    atomicAdd(&colsum[i2], L2);
}

__global__ void k_moe_loss(const float* __restrict__ colsum, float* __restrict__ loss) {
    float s = colsum[0] + colsum[1] + colsum[2] + colsum[3];
    float mean = 0.25f * s;
    float v = 0.f;
    for (int e = 0; e < 4; ++e) { float d = colsum[e] - mean; v += d * d; }
    v *= (1.0f / 3.0f);                       // ddof=1
    float cv = sqrtf(v) / mean;
    atomicAdd(loss, cv * cv);
}

// ---------------------------------------------------------------------------
// Attention for S=3, head_dim=32: one thread per (batch, head)
// ---------------------------------------------------------------------------
__global__ void k_attn(const float* __restrict__ qkv,  // [B*3, 384]
                       float* __restrict__ o,          // [B*3, 128]
                       int B)
{
    int t = blockIdx.x * blockDim.x + threadIdx.x;
    if (t >= B * NHEAD) return;
    int b = t >> 2;
    int h = t & 3;
    const float* base = qkv + (size_t)b * 3 * (3 * OUTD);
    const float scale = 0.1767766952966369f;   // 1/sqrt(32)
    float att[3][3];
    for (int qs = 0; qs < 3; ++qs) {
        float mx = -3.4e38f;
        for (int ks = 0; ks < 3; ++ks) {
            float a = 0.f;
            const float* q = base + qs * (3 * OUTD) + h * 32;
            const float* k = base + ks * (3 * OUTD) + OUTD + h * 32;
            for (int d = 0; d < 32; ++d) a += q[d] * k[d];
            a *= scale;
            att[qs][ks] = a;
            mx = fmaxf(mx, a);
        }
        float z = 0.f;
        for (int ks = 0; ks < 3; ++ks) { att[qs][ks] = expf(att[qs][ks] - mx); z += att[qs][ks]; }
        float inv = 1.0f / z;
        for (int ks = 0; ks < 3; ++ks) att[qs][ks] *= inv;
    }
    for (int qs = 0; qs < 3; ++qs) {
        float* op = o + (size_t)(b * 3 + qs) * OUTD + h * 32;
        for (int d = 0; d < 32; ++d) {
            float a = 0.f;
            for (int ks = 0; ks < 3; ++ks)
                a += att[qs][ks] * base[ks * (3 * OUTD) + 2 * OUTD + h * 32 + d];
            op[d] = a;
        }
    }
}

// x = LayerNorm(x + r) over 128 features, one wave per row (4 feats/lane)
__global__ __launch_bounds__(256)
void k_add_ln(float* __restrict__ x, const float* __restrict__ r,
              const float* __restrict__ g, const float* __restrict__ b, int rows)
{
    int row  = blockIdx.x * 8 + (threadIdx.x >> 5);
    int lane = threadIdx.x & 31;
    if (row >= rows) return;
    float v[4];
    float s = 0.f;
    for (int q = 0; q < 4; ++q) {
        int f = lane + q * 32;
        v[q] = x[(size_t)row * OUTD + f] + r[(size_t)row * OUTD + f];
        s += v[q];
    }
    for (int off = 16; off > 0; off >>= 1) s += __shfl_xor(s, off, 32);
    float mean = s * (1.0f / OUTD);
    float vs = 0.f;
    for (int q = 0; q < 4; ++q) { float d = v[q] - mean; vs += d * d; }
    for (int off = 16; off > 0; off >>= 1) vs += __shfl_xor(vs, off, 32);
    float inv = rsqrtf(vs * (1.0f / OUTD) + 1e-5f);
    for (int q = 0; q < 4; ++q) {
        int f = lane + q * 32;
        x[(size_t)row * OUTD + f] = (v[q] - mean) * inv * g[f] + b[f];
    }
}

__global__ void k_classifier(const float* __restrict__ seq,   // [B, 384]
                             const float* __restrict__ cw,    // [2, 384]
                             const float* __restrict__ cb,    // [2]
                             float* __restrict__ out, int B)
{
    int b = blockIdx.x * blockDim.x + threadIdx.x;
    if (b >= B) return;
    const float* s = seq + (size_t)b * (3 * OUTD);
    float o0 = cb[0], o1 = cb[1];
    for (int j = 0; j < 3 * OUTD; ++j) {
        o0 += s[j] * cw[j];
        o1 += s[j] * cw[3 * OUTD + j];
    }
    out[b * 2 + 0] = o0;
    out[b * 2 + 1] = o1;
}

__global__ void k_final_loss(const float* __restrict__ loss, float* __restrict__ out) {
    out[0] = loss[0];
}

// ---------------------------------------------------------------------------
// Host driver
// ---------------------------------------------------------------------------
extern "C" void kernel_launch(void* const* d_in, const int* in_sizes, int n_in,
                              void* d_out, int out_size, void* d_ws, size_t ws_size,
                              hipStream_t stream)
{
    (void)n_in; (void)out_size; (void)ws_size;

    const float* x   = (const float*)d_in[0];
    const int*   src = (const int*)d_in[1];
    const int*   dst = (const int*)d_in[2];
    const int N  = in_sizes[0] / XDIM;
    const int E  = in_sizes[1];
    const int B  = in_sizes[22] / NEXP;     // noise0 is [B,4]
    const int E2 = E + N;
    const int T  = 3 * B;                   // transformer token rows

    const float* g_lin1_w = (const float*)d_in[4];
    const float* g_lin1_b = (const float*)d_in[5];
    const float* g1_W   = (const float*)d_in[6];
    const float* g1_as  = (const float*)d_in[7];
    const float* g1_ad  = (const float*)d_in[8];
    const float* g1_b   = (const float*)d_in[9];
    const float* g2_W   = (const float*)d_in[10];
    const float* g2_as  = (const float*)d_in[11];
    const float* g2_ad  = (const float*)d_in[12];
    const float* g2_b   = (const float*)d_in[13];
    const float* g_lin2_w = (const float*)d_in[14];
    const float* g_lin2_b = (const float*)d_in[15];
    const float* t_w  = (const float*)d_in[16];
    const float* t_b  = (const float*)d_in[17];
    const float* m_w1 = (const float*)d_in[18];
    const float* m_b1 = (const float*)d_in[19];
    const float* m_w2 = (const float*)d_in[20];
    const float* m_b2 = (const float*)d_in[21];
    const float* f_qkv_w = (const float*)d_in[43];
    const float* f_qkv_b = (const float*)d_in[44];
    const float* f_out_w = (const float*)d_in[45];
    const float* f_out_b = (const float*)d_in[46];
    const float* f_ln1_g = (const float*)d_in[47];
    const float* f_ln1_b = (const float*)d_in[48];
    const float* f_ff1_w = (const float*)d_in[49];
    const float* f_ff1_b = (const float*)d_in[50];
    const float* f_ff2_w = (const float*)d_in[51];
    const float* f_ff2_b = (const float*)d_in[52];
    const float* f_ln2_g = (const float*)d_in[53];
    const float* f_ln2_b = (const float*)d_in[54];
    const float* c_w = (const float*)d_in[55];
    const float* c_b = (const float*)d_in[56];

    float* out = (float*)d_out;

    // ---- workspace allocator (16B aligned chunks) ----
    char* cur = (char*)d_ws;
    auto allocB = [&](size_t bytes) {
        char* p = cur;
        cur += (bytes + 15) & ~(size_t)15;
        return p;
    };
    auto allocF = [&](size_t n) { return (float*)allocB(n * 4); };
    auto allocH = [&](size_t n) { return (_Float16*)allocB(n * 2); };

    // f32 buffers
    float* GA = allocF((size_t)N * HID);
    float* GB = allocF((size_t)N * HID);   // contiguous after GA
    float* GC = allocF((size_t)N * HID);
    float* ASRC = allocF(N);
    float* ADST = allocF(N);
    float* EBUF = allocF(E2);
    unsigned* MMAX = (unsigned*)allocB((size_t)N * 4);
    float* ZSUM = allocF(N);
    float* WBUF = allocF((size_t)B * NEXP);
    float* CSUM = allocF(4);
    float* LOSS = allocF(4);
    float* SEQ  = allocF((size_t)T * OUTD);
    float* AO   = allocF((size_t)T * OUTD);
    float* PR   = allocF((size_t)T * OUTD);
    // f16 buffers
    _Float16* X16  = allocH((size_t)N * XPAD);   // padded x (rows 776 halves)
    _Float16* H16  = allocH((size_t)N * HID);    // graph activations
    _Float16* XG16 = allocH((size_t)B * OUTD);
    _Float16* XT16 = allocH((size_t)B * OUTD);
    _Float16* XM16 = allocH((size_t)B * OUTD);
    _Float16* M16  = allocH((size_t)B * HID);    // meta hidden
    _Float16* MH16 = allocH((size_t)B * HF);     // MoE hidden
    _Float16* S16  = allocH((size_t)T * OUTD);   // transformer f16 staging
    _Float16* FF16 = allocH((size_t)T * HF);
    _Float16* W16  = allocH((size_t)HID * XPAD); // weight panel scratch (max)
    // QKV f32 reuses GA+GB (dead by transformer time): T*384 <= 2*N*HID
    float* QKV = GA;

    const unsigned ORD_NEG_INF = 0x007FFFFFu;    // ford(-inf)

    auto cvt = [&](const float* s, _Float16* d, size_t n) {
        k_cvt<<<(int)((n + 255) / 256), 256, 0, stream>>>(s, d, (int)n);
    };
    auto zero = [&](float* p, size_t n) {
        k_zero<<<(int)((n + 255) / 256), 256, 0, stream>>>(p, (int)n);
    };
    // convert W [Nn,K] f32 -> W16 [Nn,Kp] f16 (padded), then launch GEMM
    auto gemm = [&](const _Float16* A, int lda, const float* Wf, int Nn, int K,
                    const float* bias, const float* rs, int rss,
                    float* C, _Float16* C16, int ldc, int M, int flags) {
        const int Kp = (K + 7) & ~7;
        {
            size_t n = (size_t)Nn * Kp;
            k_cvt_pad<<<(int)((n + 255) / 256), 256, 0, stream>>>(Wf, W16, Nn, K, Kp);
        }
        int tiles = ((M + 15) / 16) * ((Nn + 31) / 32);
        int blocks = (tiles + 7) / 8;
        k_gemm<<<blocks, 256, 0, stream>>>(A, lda, W16, Kp, bias, rs, rss,
                                           C, C16, ldc, M, Nn, K, flags);
    };

    // ================= input conversion =================
    {
        size_t n = (size_t)N * XPAD;
        k_cvt_pad<<<(int)((n + 255) / 256), 256, 0, stream>>>(x, X16, N, XDIM, XPAD);
    }

    // ================= graph branch =================
    // h0 = relu(x @ g_lin1_w^T + b)  -> f16 only
    gemm(X16, XPAD, g_lin1_w, HID, XDIM, g_lin1_b, nullptr, 0,
         nullptr, H16, HID, N, F_RELU);

    // ---- GAT conv 1 ----
    gemm(H16, HID, g1_W, HID, HID, nullptr, nullptr, 0,
         GB, nullptr, HID, N, 0);                               // hW (f32)
    zero(GC, (size_t)N * HID);
    k_fillu<<<(N + 255) / 256, 256, 0, stream>>>(MMAX, ORD_NEG_INF, N);
    zero(ZSUM, N);
    k_alpha<<<(N + 255) / 256, 256, 0, stream>>>(GB, g1_as, g1_ad, ASRC, ADST, N);
    k_edge_max<<<(E2 + 255) / 256, 256, 0, stream>>>(src, dst, E, E2, ASRC, ADST, EBUF, MMAX);
    k_edge_exp<<<(E2 + 255) / 256, 256, 0, stream>>>(src, dst, E, E2, EBUF, MMAX, ZSUM);
    {
        long long tt = (long long)E2 * 64;
        k_edge_scatter<<<(int)((tt + 255) / 256), 256, 0, stream>>>(src, dst, E, E2, EBUF, ZSUM, GB, GC);
    }
    k_bias_act<<<(int)(((size_t)N * HID + 255) / 256), 256, 0, stream>>>(GC, g1_b, N * HID, HID, 1);

    // ---- GAT conv 2 ----
    cvt(GC, H16, (size_t)N * HID);
    gemm(H16, HID, g2_W, HID, HID, nullptr, nullptr, 0,
         GA, nullptr, HID, N, 0);                               // hW (f32)
    zero(GB, (size_t)N * HID);
    k_fillu<<<(N + 255) / 256, 256, 0, stream>>>(MMAX, ORD_NEG_INF, N);
    zero(ZSUM, N);
    k_alpha<<<(N + 255) / 256, 256, 0, stream>>>(GA, g2_as, g2_ad, ASRC, ADST, N);
    k_edge_max<<<(E2 + 255) / 256, 256, 0, stream>>>(src, dst, E, E2, ASRC, ADST, EBUF, MMAX);
    k_edge_exp<<<(E2 + 255) / 256, 256, 0, stream>>>(src, dst, E, E2, EBUF, MMAX, ZSUM);
    {
        long long tt = (long long)E2 * 64;
        k_edge_scatter<<<(int)((tt + 255) / 256), 256, 0, stream>>>(src, dst, E, E2, EBUF, ZSUM, GA, GB);
    }
    k_bias_act<<<(int)(((size_t)N * HID + 255) / 256), 256, 0, stream>>>(GB, g2_b, N * HID, HID, 0);

    // x_g = relu(h[:B] @ g_lin2^T + b)  -> f16
    cvt(GB, H16, (size_t)B * HID);
    gemm(H16, HID, g_lin2_w, OUTD, HID, g_lin2_b, nullptr, 0,
         nullptr, XG16, OUTD, B, F_RELU);
    // x_t = relu(x[:B,:768] @ t_w^T + b) -> f16
    gemm(X16, XPAD, t_w, OUTD, IN_DIM, t_b, nullptr, 0,
         nullptr, XT16, OUTD, B, F_RELU);
    // x_m = relu(relu(x[:B,768:] @ m_w1^T + b1) @ m_w2^T + b2) -> f16
    gemm(X16 + IN_DIM, XPAD, m_w1, HID, META, m_b1, nullptr, 0,
         nullptr, M16, HID, B, F_RELU);
    gemm(M16, HID, m_w2, OUTD, HID, m_b2, nullptr, 0,
         nullptr, XM16, OUTD, B, F_RELU);

    // ================= MoE x3 -> SEQ [B,3,128] =================
    zero(LOSS, 1);
    zero(SEQ, (size_t)T * OUTD);
    const _Float16* xin[3] = {XG16, XT16, XM16};
    for (int j = 0; j < 3; ++j) {
        const float* eps     = (const float*)d_in[22 + 7 * j];
        const float* gate_w  = (const float*)d_in[23 + 7 * j];
        const float* noise_w = (const float*)d_in[24 + 7 * j];
        const float* mw1     = (const float*)d_in[25 + 7 * j];
        const float* mb1     = (const float*)d_in[26 + 7 * j];
        const float* mw2     = (const float*)d_in[27 + 7 * j];
        const float* mb2     = (const float*)d_in[28 + 7 * j];

        zero(CSUM, 4);
        k_moe_gate<<<(B + 255) / 256, 256, 0, stream>>>(xin[j], eps, gate_w, noise_w, WBUF, CSUM, B);
        k_moe_loss<<<1, 1, 0, stream>>>(CSUM, LOSS);

        for (int e = 0; e < NEXP; ++e) {
            // h1 = relu(x @ w1[e]^T + b1[e]) -> f16
            gemm(xin[j], OUTD, mw1 + (size_t)e * HF * OUTD, HF, OUTD,
                 mb1 + (size_t)e * HF, nullptr, 0,
                 nullptr, MH16, HF, B, F_RELU);
            // seq[:,j,:] += w[b,e] * (h1 @ w2[e]^T + b2[e])
            gemm(MH16, HF, mw2 + (size_t)e * OUTD * HF, OUTD, HF,
                 mb2 + (size_t)e * OUTD, WBUF + e, NEXP,
                 SEQ + j * OUTD, nullptr, 3 * OUTD, B, F_ACCUM);
        }
    }

    // ================= transformer encoder x4 =================
    for (int l = 0; l < NLAYER; ++l) {
        cvt(SEQ, S16, (size_t)T * OUTD);
        gemm(S16, OUTD, f_qkv_w + (size_t)l * 3 * OUTD * OUTD, 3 * OUTD, OUTD,
             f_qkv_b + (size_t)l * 3 * OUTD, nullptr, 0,
             QKV, nullptr, 3 * OUTD, T, 0);
        k_attn<<<(B * NHEAD + 255) / 256, 256, 0, stream>>>(QKV, AO, B);
        cvt(AO, S16, (size_t)T * OUTD);
        gemm(S16, OUTD, f_out_w + (size_t)l * OUTD * OUTD, OUTD, OUTD,
             f_out_b + (size_t)l * OUTD, nullptr, 0,
             PR, nullptr, OUTD, T, 0);
        k_add_ln<<<(T + 7) / 8, 256, 0, stream>>>(SEQ, PR,
             f_ln1_g + (size_t)l * OUTD, f_ln1_b + (size_t)l * OUTD, T);
        cvt(SEQ, S16, (size_t)T * OUTD);
        gemm(S16, OUTD, f_ff1_w + (size_t)l * HF * OUTD, HF, OUTD,
             f_ff1_b + (size_t)l * HF, nullptr, 0,
             nullptr, FF16, HF, T, F_RELU);
        gemm(FF16, HF, f_ff2_w + (size_t)l * OUTD * HF, OUTD, HF,
             f_ff2_b + (size_t)l * OUTD, nullptr, 0,
             PR, nullptr, OUTD, T, 0);
        k_add_ln<<<(T + 7) / 8, 256, 0, stream>>>(SEQ, PR,
             f_ln2_g + (size_t)l * OUTD, f_ln2_b + (size_t)l * OUTD, T);
    }

    // ================= classifier + loss =================
    k_classifier<<<(B + 255) / 256, 256, 0, stream>>>(SEQ, c_w, c_b, out, B);
    k_final_loss<<<1, 1, 0, stream>>>(LOSS, out + 2 * B);
}
